// RNN_11141145166187
// MI455X (gfx1250) — matrix operand order repaired and checked by reading, not compiled, not run on hardware
//
#include <hip/hip_runtime.h>
#include <hip/hip_bf16.h>

typedef _Float16 half_t;
typedef __attribute__((ext_vector_type(16))) _Float16 v16h;
typedef __attribute__((ext_vector_type(8)))  float    v8f;
typedef __attribute__((ext_vector_type(4)))  unsigned int u32x4;

#define B_    32
#define T_    1024
#define H_    512
#define L_    8
#define C_    1623
#define TWOH  1024
#define FOURH 2048
#define NWG   16      // workgroups per layer
#define NT    8       // 16x16 N-tiles per WG (4 gates * 32 cols / 16)
#define KSTEPS 32     // 1024 / 32
#define CPAD  1632
#define NTP   (CPAD/16)   // 102 N-tiles in projection

#define AROW  2064    // padded LDS row stride in bytes (1024 f16 + 16B pad)
#define ZOFF  (32 * AROW)            // z-buffer offset in dynamic LDS
#define SMEM_BYTES (ZOFF + 32 * 128 * 4)   // 66048 + 16384 = 82432

union FragA { v16h h; u32x4 u[2]; };

// ---------------- workspace layout (bytes) ----------------
constexpr size_t SZ_PW   = (size_t)L_*NWG*NT*KSTEPS*512*2;  // 33,554,432 packed W (f16)
constexpr size_t SZ_XBF  = (size_t)T_*B_*H_*2;              // 33,554,432 x in f16, time-major
constexpr size_t SZ_HTOP = (size_t)B_*T_*H_*2;              // 33,554,432 top-layer h (f16)
constexpr size_t SZ_PWO  = (size_t)NTP*16*512*2;            //  1,671,168 packed W_out
constexpr size_t SZ_HST  = (size_t)2*L_*B_*H_*2;            //    524,288 ping-pong h state
constexpr size_t SZ_CST  = (size_t)L_*B_*H_*4;              //    524,288 c state (f32)
constexpr size_t SZ_FLG  = (size_t)L_*T_*4;                 //     32,768 done flags

constexpr size_t OFF_PW   = 0;
constexpr size_t OFF_XBF  = OFF_PW   + SZ_PW;
constexpr size_t OFF_HTOP = OFF_XBF  + SZ_XBF;
constexpr size_t OFF_PWO  = OFF_HTOP + SZ_HTOP;
constexpr size_t OFF_HST  = OFF_PWO  + SZ_PWO;
constexpr size_t OFF_CST  = OFF_HST  + SZ_HST;
constexpr size_t OFF_FLG  = OFF_CST  + SZ_CST;

__device__ __forceinline__ float sigm(float x) { return 1.0f / (1.0f + __expf(-x)); }

// Async global -> LDS copy, 16B per lane, tracked by ASYNCcnt (gfx1250).
__device__ __forceinline__ void async_ld_b128(unsigned lds_addr, const half_t* g) {
  asm volatile("global_load_async_to_lds_b128 %0, %1, off"
               :: "v"(lds_addr), "v"(g) : "memory");
}
__device__ __forceinline__ void wait_asynccnt0() {
  asm volatile("s_wait_asynccnt 0x0" ::: "memory");
}

// ---------------- helper kernels ----------------
__global__ void zero_kernel(int* __restrict__ p, int n) {
  int i = blockIdx.x * 256 + threadIdx.x;
  if (i < n) p[i] = 0;
}

// Pack W [L, 2H, 4H] fp32 -> per-(layer, wg, ntile, kstep) WMMA B-fragments, f16.
// B-frag k order used here (contiguous, same as the A loads): lane L, element e -> K = e + (L&16?16:0), N = L&15.
__global__ void pack_w_kernel(const float* __restrict__ W, half_t* __restrict__ pw) {
  size_t idx = (size_t)blockIdx.x * 256 + threadIdx.x;
  if (idx >= (size_t)L_*NWG*NT*KSTEPS*512) return;
  int e    = (int)(idx & 15);
  int lane = (int)((idx >> 4) & 31);
  int ks   = (int)((idx >> 9) & 31);
  int nt   = (int)((idx >> 14) & 7);
  int w    = (int)((idx >> 17) & 15);
  int l    = (int)(idx >> 21);
  int K  = ks * 32 + e + ((lane & 16) ? 16 : 0);
  int zc = (nt >> 1) * 512 + w * 32 + (nt & 1) * 16 + (lane & 15);  // gate-interleaved column
  pw[idx] = (half_t)W[((size_t)l * TWOH + K) * FOURH + zc];
}

// Pack W_out [H, C] fp32 -> f16 B-fragments, N padded to 1632.
__global__ void pack_wout_kernel(const float* __restrict__ Wo, half_t* __restrict__ pwo) {
  size_t idx = (size_t)blockIdx.x * 256 + threadIdx.x;
  if (idx >= (size_t)NTP*16*512) return;
  int e    = (int)(idx & 15);
  int lane = (int)((idx >> 4) & 31);
  int ks   = (int)((idx >> 9) & 15);
  int ntp  = (int)(idx >> 13);
  int K   = ks * 32 + e + ((lane & 16) ? 16 : 0);
  int col = ntp * 16 + (lane & 15);
  pwo[idx] = (half_t)((col < C_) ? Wo[(size_t)K * C_ + col] : 0.0f);
}

// x [B, T, H] fp32 -> xbf [T, B, H] f16 (time-major rows for A-fragments)
__global__ void conv_x_kernel(const float* __restrict__ x, half_t* __restrict__ xbf) {
  size_t idx = (size_t)blockIdx.x * 256 + threadIdx.x;
  if (idx >= (size_t)T_*B_*H_) return;
  int k = (int)(idx & 511);
  int b = (int)((idx >> 9) & 31);
  int t = (int)(idx >> 14);
  xbf[idx] = (half_t)x[((size_t)b * T_ + t) * H_ + k];
}

// ---------------- persistent LSTM pipeline kernel ----------------
// grid = L_*NWG blocks of 256 threads (8 waves). Per layer-step the WG async-DMAs
// A = [inp | h(t-1)] (32 x 1024 f16) into LDS, then wave w computes N-tile w,
// M-tiles 0 and 1 of the [32 x 128] z-slice owned by this WG via WMMA.
__global__ __launch_bounds__(256) void lstm_kernel(
    const half_t* __restrict__ pw, const half_t* __restrict__ xbf,
    const float* __restrict__ bias, half_t* __restrict__ hstate,
    float* __restrict__ cstate, half_t* __restrict__ htop,
    int* __restrict__ flg) {
  extern __shared__ char smem[];                 // [32][AROW] A-stage + z-buffer
  float* zbuf = (float*)(smem + ZOFF);           // 32 x 128 f32

  const int l = blockIdx.x >> 4;
  const int w = blockIdx.x & 15;
  const int tid = threadIdx.x;
  const int wave = tid >> 5;
  const int lane = tid & 31;
  const int lane15 = lane & 15;
  const int lanehi = (lane >> 4) & 1;

  const half_t* wtile = pw + (((size_t)(l * NWG + w) * NT + wave) * KSTEPS) * 512;
  const float*  bvec  = bias + (size_t)l * FOURH;
  float*        cst   = cstate + (size_t)l * B_ * H_;
  const char*   arow0 = smem + (size_t)lane15 * AROW;   // A rows for this lane

  for (int t = 0; t < T_; ++t) {
    // ---- pipeline waits (acquire) ----
    if (tid == 0 && l > 0) {
      while (__hip_atomic_load(&flg[(l - 1) * T_ + t], __ATOMIC_ACQUIRE,
                               __HIP_MEMORY_SCOPE_AGENT) < NWG)
        __builtin_amdgcn_s_sleep(1);
    }
    if (tid == 32 && t > 0) {
      while (__hip_atomic_load(&flg[l * T_ + (t - 1)], __ATOMIC_ACQUIRE,
                               __HIP_MEMORY_SCOPE_AGENT) < NWG)
        __builtin_amdgcn_s_sleep(1);
    }
    if (tid == 64 && l < L_ - 1 && t >= 2) {
      while (__hip_atomic_load(&flg[(l + 1) * T_ + (t - 2)], __ATOMIC_ACQUIRE,
                               __HIP_MEMORY_SCOPE_AGENT) < NWG)
        __builtin_amdgcn_s_sleep(1);
    }
    __syncthreads();

    const int slot = t & 1;
    const half_t* inp = (l == 0)
        ? (xbf + (size_t)t * B_ * H_)
        : (hstate + ((size_t)slot * L_ + (l - 1)) * B_ * H_);
    const half_t* hself = hstate + ((size_t)(slot ^ 1) * L_ + l) * B_ * H_;

    // ---- async-DMA stage A = [inp | hself] into LDS (64KB, 16B/lane chunks) ----
#pragma unroll
    for (int it = 0; it < 16; ++it) {
      const int cid = (it << 8) + tid;        // 0..4095
      const int row = cid >> 7;               // 0..31
      const int ch  = cid & 127;              // 16B chunk within padded row
      const half_t* src = (ch < 64) ? (inp + row * 512 + ch * 8)
                                    : (hself + row * 512 + (ch - 64) * 8);
      const unsigned laddr = (unsigned)(size_t)(smem + row * AROW + ch * 16);
      async_ld_b128(laddr, src);
    }
    wait_asynccnt0();
    __syncthreads();

    // ---- GEMM: z-slice = A @ Wslice (A-frags from LDS, B-frags from L2) ----
    v8f acc0 = {0, 0, 0, 0, 0, 0, 0, 0};
    v8f acc1 = {0, 0, 0, 0, 0, 0, 0, 0};
#pragma unroll 4
    for (int ks = 0; ks < KSTEPS; ++ks) {
      const int kb2 = ((ks << 5) + (lanehi << 4)) << 1;  // byte offset in A row
      FragA a0, a1, bfr;
      a0.u[0] = *(const u32x4*)(arow0 + kb2);
      a0.u[1] = *(const u32x4*)(arow0 + kb2 + 16);
      a1.u[0] = *(const u32x4*)(arow0 + 16 * AROW + kb2);
      a1.u[1] = *(const u32x4*)(arow0 + 16 * AROW + kb2 + 16);
      const half_t* pb = wtile + ((size_t)ks << 9) + (lane << 4);
      if (ks + 1 < KSTEPS)
        __builtin_prefetch(pb + 512, 0, 1);   // next weight tile -> global_prefetch
      bfr.u[0] = *(const u32x4*)(pb);
      bfr.u[1] = *(const u32x4*)(pb + 8);
      acc0 = __builtin_amdgcn_wmma_f32_16x16x32_f16(false, a0.h, false, bfr.h,
                                                    (short)0, acc0, false, false);
      acc1 = __builtin_amdgcn_wmma_f32_16x16x32_f16(false, a1.h, false, bfr.h,
                                                    (short)0, acc1, false, false);
    }

    // ---- scatter accumulators to LDS z-buffer ----
    {
      const int col = (wave << 4) + lane15;
      const int rbase = lanehi << 3;
#pragma unroll
      for (int r = 0; r < 8; ++r) {
        zbuf[(rbase + r) * 128 + col]      = acc0[r];
        zbuf[(16 + rbase + r) * 128 + col] = acc1[r];
      }
    }
    __syncthreads();

    // ---- gate nonlinearity + state update ----
    half_t* hcur = hstate + ((size_t)slot * L_ + l) * B_ * H_;
#pragma unroll
    for (int k = 0; k < 4; ++k) {
      const int el = (k << 8) + tid;       // 0..1023
      const int bb = el >> 5;              // batch row
      const int cj = el & 31;              // local h column
      const int gcol = (w << 5) + cj;      // global h column
      const float zi = zbuf[bb * 128 + cj]      + bvec[gcol];
      const float zg = zbuf[bb * 128 + 32 + cj] + bvec[512 + gcol];
      const float zf = zbuf[bb * 128 + 64 + cj] + bvec[1024 + gcol];
      const float zo = zbuf[bb * 128 + 96 + cj] + bvec[1536 + gcol];
      const float cprev = cst[bb * H_ + gcol];
      const float cn = sigm(zf + 1.0f) * cprev + sigm(zi) * tanhf(zg);
      const float hn = sigm(zo) * tanhf(cn);
      cst[bb * H_ + gcol] = cn;
      hcur[bb * H_ + gcol] = (half_t)hn;
      if (l == L_ - 1) htop[((size_t)bb * T_ + t) * H_ + gcol] = (half_t)hn;
    }
    __syncthreads();

    // ---- release ----
    if (tid == 0) {
      __threadfence();
      __hip_atomic_fetch_add(&flg[l * T_ + t], 1, __ATOMIC_RELEASE,
                             __HIP_MEMORY_SCOPE_AGENT);
    }
  }
}

// ---------------- output projection: [B*T, 512] @ [512, 1623] + b_out ----------------
__global__ __launch_bounds__(256) void proj_kernel(
    const half_t* __restrict__ htop, const half_t* __restrict__ pwo,
    const float* __restrict__ bout, float* __restrict__ out) {
  const int wave = threadIdx.x >> 5;
  const int lane = threadIdx.x & 31;
  const int lane15 = lane & 15;
  const int lanehi = (lane >> 4) & 1;
  const long tile = (long)blockIdx.x * 8 + wave;
  if (tile >= (long)(B_ * T_ / 16) * NTP) return;
  const int mt = (int)(tile / NTP);
  const int ntp = (int)(tile % NTP);

  const half_t* arow = htop + (size_t)(mt * 16 + lane15) * H_;
  const half_t* wt = pwo + (size_t)ntp * 16 * 512 + (lane << 4);
  v8f acc = {0, 0, 0, 0, 0, 0, 0, 0};
#pragma unroll 4
  for (int ks = 0; ks < 16; ++ks) {
    const int kb = (ks << 5) + (lanehi << 4);
    FragA a, bfr;
    a.u[0] = *(const u32x4*)(arow + kb);
    a.u[1] = *(const u32x4*)(arow + kb + 8);
    const half_t* pb = wt + (size_t)ks * 512;
    bfr.u[0] = *(const u32x4*)(pb);
    bfr.u[1] = *(const u32x4*)(pb + 8);
    acc = __builtin_amdgcn_wmma_f32_16x16x32_f16(false, a.h, false, bfr.h,
                                                 (short)0, acc, false, false);
  }
  const int col = (ntp << 4) + lane15;
  if (col < C_) {
    const float bo = bout[col];
    const int mbase = mt * 16 + (lanehi << 3);
#pragma unroll
    for (int r = 0; r < 8; ++r)
      out[(size_t)(mbase + r) * C_ + col] = acc[r] + bo;
  }
}

// ---------------- host entry ----------------
extern "C" void kernel_launch(void* const* d_in, const int* in_sizes, int n_in,
                              void* d_out, int out_size, void* d_ws, size_t ws_size,
                              hipStream_t stream) {
  const float* x  = (const float*)d_in[0];   // [32, 1024, 512]
  const float* W  = (const float*)d_in[1];   // [8, 1024, 2048]
  const float* b  = (const float*)d_in[2];   // [8, 2048]
  const float* Wo = (const float*)d_in[3];   // [512, 1623]
  const float* bo = (const float*)d_in[4];   // [1623]

  char* ws = (char*)d_ws;
  half_t* pw   = (half_t*)(ws + OFF_PW);
  half_t* xbf  = (half_t*)(ws + OFF_XBF);
  half_t* htop = (half_t*)(ws + OFF_HTOP);
  half_t* pwo  = (half_t*)(ws + OFF_PWO);
  half_t* hst  = (half_t*)(ws + OFF_HST);
  float*  cst  = (float*)(ws + OFF_CST);
  int*    flg  = (int*)(ws + OFF_FLG);
  float*  out  = (float*)d_out;

  // zero h/c state + flags (contiguous region)
  const int nz = (int)((SZ_HST + SZ_CST + SZ_FLG) / 4);
  zero_kernel<<<(nz + 255) / 256, 256, 0, stream>>>((int*)(ws + OFF_HST), nz);

  pack_w_kernel<<<(1 << 24) / 256, 256, 0, stream>>>(W, pw);
  pack_wout_kernel<<<((int)(NTP * 16 * 512) + 255) / 256, 256, 0, stream>>>(Wo, pwo);
  conv_x_kernel<<<(1 << 24) / 256, 256, 0, stream>>>(x, xbf);

  lstm_kernel<<<L_ * NWG, 256, SMEM_BYTES, stream>>>(pw, xbf, b, hst, cst, htop, flg);

  proj_kernel<<<(B_ * T_ / 16) * NTP / 8, 256, 0, stream>>>(htop, pwo, bo, out);
}
